// domain_fusion_49890340110902
// MI455X (gfx1250) — compile-verified, matrix operand order checked by reading, not run on hardware
//
#include <hip/hip_runtime.h>
#include <stdint.h>

// Problem constants (match reference)
#define B_    32
#define C_    256
#define L_    4096
#define CM_   32
#define EPS_  1e-5f

#define LTILE 128          // L columns per workgroup
#define CSTR  264          // padded f16 stride of combined-tile rows (l-major)

typedef __attribute__((ext_vector_type(16))) _Float16 v16h;
typedef __attribute__((ext_vector_type(8)))  float    v8f;

union V16HU { v16h h; uint32_t u[8]; };

// K index of pair j (covers K, K+1) for an A/B f16 operand, lane group g (=lane>>4).
// Per CDNA5 ISA 16-bit A/B layout: lanes 0-15 hold K {0..7,16..23}, lanes 16-31 hold {8..15,24..31}.
__device__ __forceinline__ int kpair(int j, int g) {
    return ((j < 4) ? 0 : 16) + 2 * (j & 3) + 8 * g;
}

// ---------------------------------------------------------------------------
// Kernel 1: per-(b,c) max over L of (x+y)  -> gmax[B*C]
// ---------------------------------------------------------------------------
__global__ void __launch_bounds__(256) kmax_kernel(const float* __restrict__ x,
                                                   const float* __restrict__ y,
                                                   float* __restrict__ gmax) {
    __shared__ float red[256];
    const int row = blockIdx.x;                 // b*C + c
    const size_t base = (size_t)row * L_;
    float m = -3.4e38f;
    #pragma unroll
    for (int i = 0; i < L_ / 256; ++i) {
        const int l = threadIdx.x + 256 * i;
        m = fmaxf(m, x[base + l] + y[base + l]);
    }
    red[threadIdx.x] = m;
    __syncthreads();
    for (int s = 128; s > 0; s >>= 1) {
        if (threadIdx.x < s) red[threadIdx.x] = fmaxf(red[threadIdx.x], red[threadIdx.x + s]);
        __syncthreads();
    }
    if (threadIdx.x == 0) gmax[row] = red[0];
}

// ---------------------------------------------------------------------------
// Kernel 2: global branch tiny MLP per batch: gmax[b,:] -> ga[b,:]
// ---------------------------------------------------------------------------
__global__ void __launch_bounds__(256) kglobal_kernel(
        const float* __restrict__ gmax,
        const float* __restrict__ gw1, const float* __restrict__ gb1,
        const float* __restrict__ gg1, const float* __restrict__ gbe1,
        const float* __restrict__ gm1, const float* __restrict__ gv1,
        const float* __restrict__ gw2, const float* __restrict__ gb2,
        const float* __restrict__ gg2, const float* __restrict__ gbe2,
        const float* __restrict__ gm2, const float* __restrict__ gv2,
        float* __restrict__ ga) {
    __shared__ float g_s[C_];
    __shared__ float h_s[CM_];
    const int b = blockIdx.x;
    const int t = threadIdx.x;
    g_s[t] = gmax[b * C_ + t];
    __syncthreads();
    if (t < CM_) {
        float acc = gb1[t];
        for (int c = 0; c < C_; ++c) acc += gw1[t * C_ + c] * g_s[c];
        const float inv = gg1[t] * rsqrtf(gv1[t] + EPS_);
        h_s[t] = fmaxf(acc * inv + (gbe1[t] - gm1[t] * inv), 0.f);
    }
    __syncthreads();
    float acc = gb2[t];
    #pragma unroll
    for (int k = 0; k < CM_; ++k) acc += gw2[t * CM_ + k] * h_s[k];
    const float inv = gg2[t] * rsqrtf(gv2[t] + EPS_);
    ga[b * C_ + t] = acc * inv + (gbe2[t] - gm2[t] * inv);
}

// ---------------------------------------------------------------------------
// Kernel 3: fused local branch (two f16 WMMA GEMMs) + sigmoid blend epilogue.
// One block = one batch x 128 L columns. 8 waves, each owning 16 columns.
// Phases: [stage combined -> LDS] | [GEMM1] | barrier | [GEMM2, att -> LDS
// (overlaid on dead combined tile)] | barrier | [coalesced float4 blend].
// ---------------------------------------------------------------------------
__global__ void __launch_bounds__(256, 1) kmain_kernel(
        const float* __restrict__ x, const float* __restrict__ y,
        const float* __restrict__ lw1, const float* __restrict__ lb1,
        const float* __restrict__ lg1, const float* __restrict__ lbe1,
        const float* __restrict__ lm1, const float* __restrict__ lv1,
        const float* __restrict__ lw2, const float* __restrict__ lb2,
        const float* __restrict__ lg2, const float* __restrict__ lbe2,
        const float* __restrict__ lm2, const float* __restrict__ lv2,
        const float* __restrict__ ga_g, float* __restrict__ out) {
    extern __shared__ char smem[];
    _Float16* comb = (_Float16*)smem;                          // [LTILE][CSTR], l-major
    _Float16* att16 = comb;                                    // overlay: [C_][LTILE] after GEMM1
    _Float16* w1s  = (_Float16*)(smem + (size_t)LTILE * CSTR * 2);  // [32][256], BN-folded
    _Float16* w2s  = w1s + 32 * 256;                           // [256][32], BN-folded
    float*    b1s  = (float*)(w2s + 256 * 32);                 // [32]  folded bias1
    float*    b2s  = b1s + 32;                                 // [256] folded bias2
    float*    gas  = b2s + 256;                                // [256] global-branch add

    const int b    = blockIdx.x >> 5;          // 32 L-tiles per batch
    const int l0   = (blockIdx.x & 31) * LTILE;
    const int t    = threadIdx.x;
    const int lane = t & 31;
    const int wave = t >> 5;

    // ---- Fold BN into f16 weights, stage biases / ga row -------------------
    #pragma unroll 4
    for (int i = 0; i < 32; ++i) {
        const int idx = t * 32 + i;            // 0..8191
        const int o1 = idx >> 8;               // lw1: [32][256]
        const float inv1 = lg1[o1] * rsqrtf(lv1[o1] + EPS_);
        w1s[idx] = (_Float16)(lw1[idx] * inv1);
        const int o2 = idx >> 5;               // lw2: [256][32]
        const float inv2 = lg2[o2] * rsqrtf(lv2[o2] + EPS_);
        w2s[idx] = (_Float16)(lw2[idx] * inv2);
    }
    if (t < 32) {
        const float inv = lg1[t] * rsqrtf(lv1[t] + EPS_);
        b1s[t] = lb1[t] * inv + lbe1[t] - lm1[t] * inv;
    }
    {
        const float inv = lg2[t] * rsqrtf(lv2[t] + EPS_);
        b2s[t] = lb2[t] * inv + lbe2[t] - lm2[t] * inv;
        gas[t] = ga_g[b * C_ + t];
    }

    // ---- Stage combined = x+y as f16 into LDS (l-major for u32 K-pairs) ----
    // Wave w loads channels [w*32, w*32+32); lanes cover the 128 columns (float4).
    #pragma unroll 4
    for (int i = 0; i < 32; ++i) {
        const int c = wave * 32 + i;
        const size_t base = ((size_t)b * C_ + c) * L_ + l0 + lane * 4;
        const float4 xv = *(const float4*)(x + base);
        const float4 yv = *(const float4*)(y + base);
        const int lr = lane * 4;
        comb[(lr + 0) * CSTR + c] = (_Float16)(xv.x + yv.x);
        comb[(lr + 1) * CSTR + c] = (_Float16)(xv.y + yv.y);
        comb[(lr + 2) * CSTR + c] = (_Float16)(xv.z + yv.z);
        comb[(lr + 3) * CSTR + c] = (_Float16)(xv.w + yv.w);
    }
    __syncthreads();

    // ---- Compute: wave owns 16 columns -------------------------------------
    const int n = lane & 15;                   // column within wave tile / M row for A
    const int g = lane >> 4;                   // lane group -> K sub-pattern
    const int lcol = wave * 16 + n;            // column within block tile

    // GEMM1: h[32 x 16] = W1'[32 x 256] * comb[256 x 16], fp32 accum
    v8f acc0 = {};
    v8f acc1 = {};
    #pragma unroll
    for (int kt = 0; kt < 8; ++kt) {
        V16HU bu, a0, a1;
        #pragma unroll
        for (int j = 0; j < 8; ++j) {
            const int kb = kt * 32 + kpair(j, g);
            bu.u[j] = *(const uint32_t*)&comb[lcol * CSTR + kb];
            a0.u[j] = *(const uint32_t*)&w1s[n * 256 + kb];
            a1.u[j] = *(const uint32_t*)&w1s[(16 + n) * 256 + kb];
        }
        acc0 = __builtin_amdgcn_wmma_f32_16x16x32_f16(false, a0.h, false, bu.h,
                                                      (short)0, acc0, false, false);
        acc1 = __builtin_amdgcn_wmma_f32_16x16x32_f16(false, a1.h, false, bu.h,
                                                      (short)0, acc1, false, false);
    }

    // bias + ReLU + cvt: C/D layout of GEMM1 == B-operand layout of GEMM2 (no shuffle)
    V16HU h2;
    #pragma unroll
    for (int e = 0; e < 16; ++e) {
        const int m = (e & 7) + 8 * g + ((e >= 8) ? 16 : 0);
        const float v = ((e < 8) ? acc0[e] : acc1[e - 8]) + b1s[m];
        h2.h[e] = (_Float16)fmaxf(v, 0.f);
    }

    // combined tile is now dead everywhere -> safe to overlay att16 after this barrier
    __syncthreads();

    // GEMM2 (K=32, one WMMA per 16-row output tile); att -> LDS (f16, c-major)
    #pragma unroll 4
    for (int mt = 0; mt < 16; ++mt) {
        V16HU a2;
        #pragma unroll
        for (int j = 0; j < 8; ++j) {
            const int kb = kpair(j, g);
            a2.u[j] = *(const uint32_t*)&w2s[(mt * 16 + n) * 32 + kb];
        }
        v8f acc2 = {};
        acc2 = __builtin_amdgcn_wmma_f32_16x16x32_f16(false, a2.h, false, h2.h,
                                                      (short)0, acc2, false, false);
        #pragma unroll
        for (int r = 0; r < 8; ++r) {
            const int m = mt * 16 + r + 8 * g;
            const float z   = acc2[r] + b2s[m] + gas[m];
            const float att = 1.f / (1.f + __expf(-z));
            att16[m * LTILE + lcol] = (_Float16)att;
        }
    }
    __syncthreads();

    // ---- Coalesced blend epilogue: float4 x/y loads (L2-hot), float4 stores -
    #pragma unroll 4
    for (int i = 0; i < 32; ++i) {
        const int c = wave * 32 + i;
        const size_t base = ((size_t)b * C_ + c) * L_ + l0 + lane * 4;
        const float4 xv = *(const float4*)(x + base);
        const float4 yv = *(const float4*)(y + base);
        const int lr = lane * 4;
        const float a0f = (float)att16[c * LTILE + lr + 0];
        const float a1f = (float)att16[c * LTILE + lr + 1];
        const float a2f = (float)att16[c * LTILE + lr + 2];
        const float a3f = (float)att16[c * LTILE + lr + 3];
        float4 ov;
        ov.x = 2.f * xv.x * a0f + 2.f * yv.x * (1.f - a0f);
        ov.y = 2.f * xv.y * a1f + 2.f * yv.y * (1.f - a1f);
        ov.z = 2.f * xv.z * a2f + 2.f * yv.z * (1.f - a2f);
        ov.w = 2.f * xv.w * a3f + 2.f * yv.w * (1.f - a3f);
        *(float4*)(out + base) = ov;
    }
}

// ---------------------------------------------------------------------------
extern "C" void kernel_launch(void* const* d_in, const int* in_sizes, int n_in,
                              void* d_out, int out_size, void* d_ws, size_t ws_size,
                              hipStream_t stream) {
    (void)in_sizes; (void)n_in; (void)out_size; (void)ws_size;

    const float* x    = (const float*)d_in[0];
    const float* y    = (const float*)d_in[1];
    const float* lw1  = (const float*)d_in[2];
    const float* lb1  = (const float*)d_in[3];
    const float* lg1  = (const float*)d_in[4];
    const float* lbe1 = (const float*)d_in[5];
    const float* lm1  = (const float*)d_in[6];
    const float* lv1  = (const float*)d_in[7];
    const float* lw2  = (const float*)d_in[8];
    const float* lb2  = (const float*)d_in[9];
    const float* lg2  = (const float*)d_in[10];
    const float* lbe2 = (const float*)d_in[11];
    const float* lm2  = (const float*)d_in[12];
    const float* lv2  = (const float*)d_in[13];
    const float* gw1  = (const float*)d_in[14];
    const float* gb1  = (const float*)d_in[15];
    const float* gg1  = (const float*)d_in[16];
    const float* gbe1 = (const float*)d_in[17];
    const float* gm1  = (const float*)d_in[18];
    const float* gv1  = (const float*)d_in[19];
    const float* gw2  = (const float*)d_in[20];
    const float* gb2  = (const float*)d_in[21];
    const float* gg2  = (const float*)d_in[22];
    const float* gbe2 = (const float*)d_in[23];
    const float* gm2  = (const float*)d_in[24];
    const float* gv2  = (const float*)d_in[25];
    float* out = (float*)d_out;

    float* gmax = (float*)d_ws;           // B*C floats
    float* ga   = gmax + B_ * C_;         // B*C floats

    kmax_kernel<<<B_ * C_, 256, 0, stream>>>(x, y, gmax);
    kglobal_kernel<<<B_, 256, 0, stream>>>(gmax, gw1, gb1, gg1, gbe1, gm1, gv1,
                                           gw2, gb2, gg2, gbe2, gm2, gv2, ga);

    const size_t shmem = (size_t)LTILE * CSTR * 2   // combined tile (f16) / att overlay
                       + 32 * 256 * 2               // w1 (f16)
                       + 256 * 32 * 2               // w2 (f16)
                       + (32 + 256 + 256) * 4;      // b1, b2, ga (f32)
    kmain_kernel<<<B_ * (L_ / LTILE), 256, shmem, stream>>>(
        x, y, lw1, lb1, lg1, lbe1, lm1, lv1,
        lw2, lb2, lg2, lbe2, lm2, lv2, ga, out);
}
